// Separable_NL_Spatial_1992864825603
// MI455X (gfx1250) — compile-verified, matrix operand order checked by reading, not run on hardware
//
#include <hip/hip_runtime.h>
#include <hip/hip_bf16.h>
#include <math.h>

#define B_   4
#define N_   5
#define C_   64
#define HWX  4096
#define NCX  320     // N_*C_
#define TQ   32      // queries per workgroup
#define TK   64      // keys per chunk
#define NCHUNK (HWX / TK)

typedef __bf16 bf16;
typedef __attribute__((ext_vector_type(16))) __bf16 v16bf;
typedef __attribute__((ext_vector_type(8)))  float  v8f;

// ---- CDNA5 async global->LDS copy (16B per lane), tracked by ASYNCcnt ----
__device__ __forceinline__ void async_cp16(void* lds, const void* g) {
  unsigned l32 = (unsigned)(unsigned long long)(lds);   // low 32 bits of flat LDS addr == LDS byte address
  unsigned long long g64 = (unsigned long long)(g);
  asm volatile("global_load_async_to_lds_b128 %0, %1, off"
               :: "v"(l32), "v"(g64) : "memory");
}
__device__ __forceinline__ void wait_async0()  { asm volatile("s_wait_asynccnt 0"  ::: "memory"); }
__device__ __forceinline__ void wait_async20() { asm volatile("s_wait_asynccnt 20" ::: "memory"); }

__device__ __forceinline__ v8f vzero8() {
  v8f z;
#pragma unroll
  for (int i = 0; i < 8; ++i) z[i] = 0.0f;
  return z;
}

#define WMMA_BF16(A, B, C, RA) \
  __builtin_amdgcn_wmma_f32_16x16x32_bf16(false, (A), false, (B), (short)0, (C), (RA), false)

// ---------------- projection: channel-mix over N (5x5), 3 weight sets ----------------
// Q,K -> [B][HW][NC] bf16 ; V -> [B][NC][HW] bf16
__global__ void pre_mixN(const float* __restrict__ x,
                         const float* __restrict__ WA, const float* __restrict__ bA,
                         const float* __restrict__ WB, const float* __restrict__ bB,
                         const float* __restrict__ WD, const float* __restrict__ bD,
                         bf16* __restrict__ Q, bf16* __restrict__ K, bf16* __restrict__ V) {
  int tid = blockIdx.x * 256 + threadIdx.x;        // B_*C_*HWX threads
  int hw  = tid & (HWX - 1);
  int c   = (tid >> 12) & (C_ - 1);
  int b   = tid >> 18;
  float xv[N_];
#pragma unroll
  for (int i = 0; i < N_; ++i)
    xv[i] = x[(((size_t)b * N_ + i) * C_ + c) * HWX + hw];
#pragma unroll
  for (int n = 0; n < N_; ++n) {
    float q = bA[n], k = bB[n], v = bD[n];
#pragma unroll
    for (int i = 0; i < N_; ++i) {
      q += WA[n * N_ + i] * xv[i];
      k += WB[n * N_ + i] * xv[i];
      v += WD[n * N_ + i] * xv[i];
    }
    size_t qidx = ((size_t)b * HWX + hw) * NCX + n * C_ + c;
    Q[qidx] = (bf16)q;
    K[qidx] = (bf16)k;
    V[((size_t)b * NCX + n * C_ + c) * HWX + hw] = (bf16)v;
  }
}

// ---------------- projection: channel-mix over C (64x64), 3 weight sets ----------------
__global__ void pre_mixC(const float* __restrict__ x,
                         const float* __restrict__ WA, const float* __restrict__ bA,
                         const float* __restrict__ WB, const float* __restrict__ bB,
                         const float* __restrict__ WD, const float* __restrict__ bD,
                         bf16* __restrict__ Q, bf16* __restrict__ K, bf16* __restrict__ V) {
  __shared__ bf16 xs[C_][256];
  const int t   = threadIdx.x;
  const int hw0 = (blockIdx.x & 15) << 8;          // 16 tiles of 256 over HW
  const int n   = (blockIdx.x >> 4) % N_;
  const int b   = blockIdx.x / (16 * N_);
  const float* xb = x + ((size_t)b * N_ + n) * C_ * HWX;
  for (int i = 0; i < C_; ++i)
    xs[i][t] = (bf16)xb[(size_t)i * HWX + hw0 + t];
  __syncthreads();
  const int hw = hw0 + t;
  for (int o = 0; o < C_; ++o) {
    float q = bA[o], k = bB[o], v = bD[o];
#pragma unroll 8
    for (int i = 0; i < C_; ++i) {
      float xvf = (float)xs[i][t];
      q += WA[o * C_ + i] * xvf;
      k += WB[o * C_ + i] * xvf;
      v += WD[o * C_ + i] * xvf;
    }
    size_t qidx = ((size_t)b * HWX + hw) * NCX + n * C_ + o;
    Q[qidx] = (bf16)q;
    K[qidx] = (bf16)k;
    V[((size_t)b * NCX + n * C_ + o) * HWX + hw] = (bf16)v;
  }
}

// ---------------- flash attention: out = (mode? out : x) + softmax(Q K^T / s) V ----------------
// 256 threads = 8 waves. Wave (qi,dj): qi picks the 16-query tile, dj picks an 80-wide
// slice of the 320 output columns. K/V chunks double-buffered in LDS via async-to-LDS.
__global__ void __launch_bounds__(256)
attn(const bf16* __restrict__ Q, const bf16* __restrict__ K,
     const bf16* __restrict__ V, const float* __restrict__ scale,
     const float* __restrict__ xin, float* __restrict__ out, int mode) {
  extern __shared__ char smemraw[];
  bf16* Qs  = (bf16*)smemraw;          // [TQ][NCX]
  bf16* Ks0 = Qs  + TQ * NCX;          // [TK][NCX] buffer 0
  bf16* Ks1 = Ks0 + TK * NCX;          // [TK][NCX] buffer 1
  bf16* Vs0 = Ks1 + TK * NCX;          // [NCX][TK] buffer 0
  bf16* Vs1 = Vs0 + NCX * TK;          // [NCX][TK] buffer 1
  bf16* Ps  = Vs1 + NCX * TK;          // [8 waves][16][TK]

  const int tid  = threadIdx.x;
  const int wid  = tid >> 5;
  const int lane = tid & 31;
  const int h    = lane >> 4;          // lane half: rows h*8..h*8+7 of each 16x16 tile
  const int nl   = lane & 15;
  const int qi   = wid >> 2;
  const int dj   = wid & 3;

  const int b  = blockIdx.x >> 7;      // HWX/TQ = 128 blocks per batch
  const int q0 = (blockIdx.x & 127) * TQ;

  const bf16* Qg = Q + (size_t)b * HWX * NCX;
  const bf16* Kg = K + (size_t)b * HWX * NCX;
  const bf16* Vg = V + (size_t)b * NCX * HWX;

  // stage Q block (async, 5 instr/wave)
  for (int g = tid; g < (TQ * NCX) / 8; g += 256) {
    int e = g * 8, row = e / NCX, col = e % NCX;
    async_cp16(Qs + row * NCX + col, Qg + (size_t)(q0 + row) * NCX + col);
  }
  // issue chunk 0 K/V (20 instr/wave)
  {
    for (int g = tid; g < (TK * NCX) / 8; g += 256) {
      int e = g * 8, row = e / NCX, col = e % NCX;
      async_cp16(Ks0 + row * NCX + col, Kg + (size_t)row * NCX + col);
    }
    for (int g = tid; g < (NCX * TK) / 8; g += 256) {
      int e = g * 8, row = e / TK, col = e % TK;
      async_cp16(Vs0 + row * TK + col, Vg + (size_t)row * HWX + col);
    }
  }

  const float invs = 1.0f / scale[0];

  float m8[8], l8[8];
  v8f acc[5];
#pragma unroll
  for (int r = 0; r < 8; ++r) { m8[r] = -INFINITY; l8[r] = 0.0f; }
#pragma unroll
  for (int o = 0; o < 5; ++o) acc[o] = vzero8();

  for (int i = 0; i < NCHUNK; ++i) {
    const bf16* Kb = (i & 1) ? Ks1 : Ks0;
    const bf16* Vb = (i & 1) ? Vs1 : Vs0;

    // prefetch chunk i+1 into the other buffer (20 instr/wave)
    if (i + 1 < NCHUNK) {
      bf16* Kn = (i & 1) ? Ks0 : Ks1;
      bf16* Vn = (i & 1) ? Vs0 : Vs1;
      const int k1 = (i + 1) * TK;
      for (int g = tid; g < (TK * NCX) / 8; g += 256) {
        int e = g * 8, row = e / NCX, col = e % NCX;
        async_cp16(Kn + row * NCX + col, Kg + (size_t)(k1 + row) * NCX + col);
      }
      for (int g = tid; g < (NCX * TK) / 8; g += 256) {
        int e = g * 8, row = e / TK, col = e % TK;
        async_cp16(Vn + row * TK + col, Vg + (size_t)row * HWX + k1 + col);
      }
      wait_async20();   // everything except the 20 just-issued has landed (in-order)
    } else {
      wait_async0();
    }
    __syncthreads();    // chunk i fully visible in LDS for all waves

    // ---- S tile: 16 queries x 64 keys, contraction over NCX in chunks of 32 ----
    v8f s[4];
#pragma unroll
    for (int t = 0; t < 4; ++t) s[t] = vzero8();
#pragma unroll
    for (int c = 0; c < NCX / 32; ++c) {
      v16bf a  = *(const v16bf*)(Qs + (qi * 16 + nl) * NCX + c * 32 + h * 16);
      v16bf b0 = *(const v16bf*)(Kb + ( 0 + nl) * NCX + c * 32 + h * 16);
      v16bf b1 = *(const v16bf*)(Kb + (16 + nl) * NCX + c * 32 + h * 16);
      v16bf b2 = *(const v16bf*)(Kb + (32 + nl) * NCX + c * 32 + h * 16);
      v16bf b3 = *(const v16bf*)(Kb + (48 + nl) * NCX + c * 32 + h * 16);
      s[0] = WMMA_BF16(a, b0, s[0], false);
      s[1] = WMMA_BF16(a, b1, s[1], true);   // reuse cached A operand
      s[2] = WMMA_BF16(a, b2, s[2], true);
      s[3] = WMMA_BF16(a, b3, s[3], true);
    }
#pragma unroll
    for (int t = 0; t < 4; ++t)
#pragma unroll
      for (int r = 0; r < 8; ++r) s[t][r] *= invs;

    // ---- online softmax: rows live across 16-lane halves, reduce with shfl_xor ----
    float mn[8];
#pragma unroll
    for (int r = 0; r < 8; ++r) {
      float v = fmaxf(fmaxf(s[0][r], s[1][r]), fmaxf(s[2][r], s[3][r]));
      v = fmaxf(v, __shfl_xor(v, 1, 32));
      v = fmaxf(v, __shfl_xor(v, 2, 32));
      v = fmaxf(v, __shfl_xor(v, 4, 32));
      v = fmaxf(v, __shfl_xor(v, 8, 32));
      mn[r] = fmaxf(m8[r], v);
    }
#pragma unroll
    for (int r = 0; r < 8; ++r) {
      float rs = 0.0f;
#pragma unroll
      for (int t = 0; t < 4; ++t) {
        float e = __expf(s[t][r] - mn[r]);
        rs += e;
        Ps[wid * (16 * TK) + (h * 8 + r) * TK + t * 16 + nl] = (bf16)e;
      }
      rs += __shfl_xor(rs, 1, 32);
      rs += __shfl_xor(rs, 2, 32);
      rs += __shfl_xor(rs, 4, 32);
      rs += __shfl_xor(rs, 8, 32);
      float alpha = __expf(m8[r] - mn[r]);
      l8[r] = l8[r] * alpha + rs;
      m8[r] = mn[r];
#pragma unroll
      for (int o = 0; o < 5; ++o) acc[o][r] *= alpha;
    }

    // ---- O += P x V_slice (contraction over 64 keys, 2 chunks of 32) ----
#pragma unroll
    for (int c2 = 0; c2 < 2; ++c2) {
      v16bf pa = *(const v16bf*)(Ps + wid * (16 * TK) + nl * TK + c2 * 32 + h * 16);
      v16bf v0 = *(const v16bf*)(Vb + (dj * 80 +  0 + nl) * TK + c2 * 32 + h * 16);
      v16bf v1 = *(const v16bf*)(Vb + (dj * 80 + 16 + nl) * TK + c2 * 32 + h * 16);
      v16bf v2 = *(const v16bf*)(Vb + (dj * 80 + 32 + nl) * TK + c2 * 32 + h * 16);
      v16bf v3 = *(const v16bf*)(Vb + (dj * 80 + 48 + nl) * TK + c2 * 32 + h * 16);
      v16bf v4 = *(const v16bf*)(Vb + (dj * 80 + 64 + nl) * TK + c2 * 32 + h * 16);
      acc[0] = WMMA_BF16(pa, v0, acc[0], false);
      acc[1] = WMMA_BF16(pa, v1, acc[1], true);  // reuse cached A operand
      acc[2] = WMMA_BF16(pa, v2, acc[2], true);
      acc[3] = WMMA_BF16(pa, v3, acc[3], true);
      acc[4] = WMMA_BF16(pa, v4, acc[4], true);
    }
    __syncthreads();  // all waves done reading this buffer; next iter may overwrite it
  }

  // ---- epilogue: out[b][nc][hw] = base + acc/l ----
#pragma unroll
  for (int o = 0; o < 5; ++o) {
    int nc = dj * 80 + o * 16 + nl;
#pragma unroll
    for (int r = 0; r < 8; ++r) {
      int q = q0 + qi * 16 + h * 8 + r;
      size_t idx = ((size_t)b * NCX + nc) * HWX + q;
      float val = acc[o][r] / l8[r];
      out[idx] = mode ? (out[idx] + val) : (xin[idx] + val);
    }
  }
}

extern "C" void kernel_launch(void* const* d_in, const int* in_sizes, int n_in,
                              void* d_out, int out_size, void* d_ws, size_t ws_size,
                              hipStream_t stream) {
  (void)in_sizes; (void)n_in; (void)out_size; (void)ws_size;
  const float* x     = (const float*)d_in[0];
  const float* WA1T  = (const float*)d_in[1];
  const float* bA1T  = (const float*)d_in[2];
  const float* WA1C  = (const float*)d_in[3];
  const float* bA1C  = (const float*)d_in[4];
  const float* WB1T  = (const float*)d_in[5];
  const float* bB1T  = (const float*)d_in[6];
  const float* WB1C  = (const float*)d_in[7];
  const float* bB1C  = (const float*)d_in[8];
  const float* WD1T  = (const float*)d_in[9];
  const float* bD1T  = (const float*)d_in[10];
  const float* WD1C  = (const float*)d_in[11];
  const float* bD1C  = (const float*)d_in[12];
  const float* scale = (const float*)d_in[13];
  float* out = (float*)d_out;

  size_t bufElems = (size_t)B_ * HWX * NCX;   // 5,242,880
  bf16* Q = (bf16*)d_ws;
  bf16* K = Q + bufElems;
  bf16* V = K + bufElems;

  const int ldsBytes = (TQ * NCX + 2 * TK * NCX + 2 * NCX * TK + 8 * 16 * TK) * (int)sizeof(bf16);

  // Branch T: out = x + E1T
  pre_mixN<<<(B_ * C_ * HWX) / 256, 256, 0, stream>>>(x, WA1T, bA1T, WB1T, bB1T, WD1T, bD1T, Q, K, V);
  attn<<<B_ * (HWX / TQ), 256, ldsBytes, stream>>>(Q, K, V, scale, x, out, 0);
  // Branch C: out += E1C (stream-ordered after branch T)
  pre_mixC<<<B_ * N_ * (HWX / 256), 256, 0, stream>>>(x, WA1C, bA1C, WB1C, bB1C, WD1C, bD1C, Q, K, V);
  attn<<<B_ * (HWX / TQ), 256, ldsBytes, stream>>>(Q, K, V, scale, x, out, 1);
}
